// QMPN_54838142435822
// MI455X (gfx1250) — matrix operand order repaired and checked by reading, not compiled, not run on hardware
//
#include <hip/hip_runtime.h>
#include <hip/hip_bf16.h>
#include <math.h>

#define DD 256
#define BB 8
#define TT 32
#define BETA_C 0.8f

typedef __attribute__((ext_vector_type(16))) __bf16 v16bf;
typedef __attribute__((ext_vector_type(8)))  __bf16 v8bf;
typedef __attribute__((ext_vector_type(8)))  float  v8f;

// ---------------- WMMA fragment helpers (CDNA5 16x16x32 bf16 layouts) ----------------
// A 16x32 (MxK): lanes 0-15 -> M=lane, elems 0..7 = K 0..7,  elems 8..15 = K 16..23
//                lanes 16-31 -> M=lane-16, elems 0..7 = K 8..15, elems 8..15 = K 24..31
// All A sources are row-major with contiguous K -> two aligned vector loads.

__device__ inline v16bf load_a(const float* __restrict__ X, int m0, int k0, int lane) {
  int m  = m0 + (lane & 15);
  int kh = (lane & 16) ? 8 : 0;
  const float* p = X + (size_t)m * DD + k0 + kh;      // 32B aligned
  v8f lo = *(const v8f*)p;                            // K = k0+kh .. +7
  v8f hi = *(const v8f*)(p + 16);                     // K = k0+kh+16 .. +23
  v16bf a;
#pragma unroll
  for (int e = 0; e < 8; ++e) { a[e] = (__bf16)lo[e]; a[e + 8] = (__bf16)hi[e]; }
  return a;
}

__device__ inline v16bf load_a(const __bf16* __restrict__ X, int m0, int k0, int lane) {
  int m  = m0 + (lane & 15);
  int kh = (lane & 16) ? 8 : 0;
  const __bf16* p = X + (size_t)m * DD + k0 + kh;     // 16B aligned
  v8bf lo = *(const v8bf*)p;
  v8bf hi = *(const v8bf*)(p + 16);
  v16bf a;
#pragma unroll
  for (int e = 0; e < 8; ++e) { a[e] = lo[e]; a[e + 8] = hi[e]; }
  return a;
}

// B 32x16 (KxN): lanes 0-15 -> N=lane, elems 0..7 = K 0..7, elems 8..15 = K 16..23;
// lanes 16-31 -> N=lane-16, elems 0..7 = K 8..15, elems 8..15 = K 24..31.
// Source is stored TRANSPOSED (src[n][k]) so each fragment is two aligned 16B loads.
__device__ inline v16bf load_bT(const __bf16* __restrict__ Yt, int k0, int n0, int lane) {
  int n  = n0 + (lane & 15);
  int kl = (lane & 16) ? 8 : 0;
  const __bf16* p = Yt + (size_t)n * DD + k0 + kl;   // 16B aligned
  v8bf lo = *(const v8bf*)p;
  v8bf hi = *(const v8bf*)(p + 16);
  v16bf r;
#pragma unroll
  for (int e = 0; e < 8; ++e) { r[e] = lo[e]; r[e + 8] = hi[e]; }
  return r;
}

__device__ inline v16bf bfneg(v16bf x) {
  v16bf y;
#pragma unroll
  for (int e = 0; e < 16; ++e) y[e] = (__bf16)(-(float)x[e]);
  return y;
}

__device__ inline v8f wmma_bf16(v16bf a, v16bf b, v8f c) {
  // (neg_a, A, neg_b, B, c_mod, C, reuse_a, reuse_b); NEG bits must be 0 for bf16
  return __builtin_amdgcn_wmma_f32_16x16x32_bf16(false, a, false, b, (short)0, c, false, false);
}

// ---------------- Complex GEMM: C = (Xr+iXi) * (Yr+iYi)^logical ----------------
// Y source is always bf16 stored transposed (Yt[n][k] = Ylogical[k][n]); CONJY flips Yi.
// One wave computes a 16x64 complex strip (4 n-subtiles) -> A fragments reused 4x,
// 16 v_wmma per k-step. Grid: (4 n-groups, 16 m-tiles, batch).
template<typename TX, bool CONJY>
__global__ __launch_bounds__(32)
void cgemm256(const TX* __restrict__ Xr, const TX* __restrict__ Xi, long sx,
              const __bf16* __restrict__ Yr, const __bf16* __restrict__ Yi, long sy,
              float* __restrict__ Cr, float* __restrict__ Ci, long sc) {
  const int lane = threadIdx.x;
  const int n0 = blockIdx.x * 64;
  const int m0 = blockIdx.y * 16;
  const int b  = blockIdx.z;
  const TX* xr = Xr + (size_t)b * sx;
  const TX* xi = Xi + (size_t)b * sx;
  const __bf16* yr = Yr + (size_t)b * sy;
  const __bf16* yi = Yi + (size_t)b * sy;
  v8f cr[4] = {}; v8f ci[4] = {};
  for (int k0 = 0; k0 < DD; k0 += 32) {
    v16bf ar = load_a(xr, m0, k0, lane);
    v16bf ai = load_a(xi, m0, k0, lane);
    v16bf an = bfneg(ai);
#pragma unroll
    for (int nt = 0; nt < 4; ++nt) {
      v16bf br = load_bT(yr, k0, n0 + nt * 16, lane);
      v16bf bi = load_bT(yi, k0, n0 + nt * 16, lane);
      if (CONJY) bi = bfneg(bi);
      cr[nt] = wmma_bf16(ar, br, cr[nt]);   // + Xr*Yr
      cr[nt] = wmma_bf16(an, bi, cr[nt]);   // - Xi*Yi
      ci[nt] = wmma_bf16(ar, bi, ci[nt]);   // + Xr*Yi
      ci[nt] = wmma_bf16(ai, br, ci[nt]);   // + Xi*Yr
    }
  }
  // C/D layout: VGPR v -> M = v (lanes 0-15) / v+8 (lanes 16-31); N = lane&15
  const int n  = lane & 15;
  const int mh = (lane & 16) ? 8 : 0;
#pragma unroll
  for (int nt = 0; nt < 4; ++nt) {
#pragma unroll
    for (int v = 0; v < 8; ++v) {
      size_t off = (size_t)b * sc + (size_t)(m0 + mh + v) * DD + n0 + nt * 16 + n;
      Cr[off] = cr[nt][v];
      Ci[off] = ci[nt][v];
    }
  }
}

// ---------------- Measurement: probs_k = Re<m_k| rho |m_k> via 4 WMMA GEMMs + diag-dot -------
// h2 state comes in transposed bf16 (h2t[j][i] = h2[i][j]) -> TRANS loads give exact B frags.
// MrT/MiT are transposed bf16 projector rows so the diag-dot reads are contiguous v8bf loads.
__global__ __launch_bounds__(32)
void meas_k(const __bf16* __restrict__ h2tr, const __bf16* __restrict__ h2ti,
            const __bf16* __restrict__ Mr, const __bf16* __restrict__ Mi,
            const __bf16* __restrict__ MrT, const __bf16* __restrict__ MiT,
            float* __restrict__ probs, int t) {
  const int lane = threadIdx.x;
  const int k0 = blockIdx.x * 16;
  const int b  = blockIdx.y;
  const __bf16* hr = h2tr + (size_t)b * DD * DD;
  const __bf16* hi = h2ti + (size_t)b * DD * DD;
  // hoist the Mr/Mi A-fragments: shared across every j-tile
  v16bf amr[8], ami[8];
#pragma unroll
  for (int s = 0; s < 8; ++s) {
    amr[s] = load_a(Mr, k0, s * 32, lane);
    ami[s] = load_a(Mi, k0, s * 32, lane);
  }
  float ps[8];
#pragma unroll
  for (int v = 0; v < 8; ++v) ps[v] = 0.f;
  const int n  = lane & 15;
  const int mh = (lane & 16) ? 8 : 0;
  for (int j0 = 0; j0 < DD; j0 += 16) {
    v8f v1 = {}, v2 = {}, v3 = {}, v4 = {};
#pragma unroll
    for (int s = 0; s < 8; ++s) {
      v16bf bhr = load_bT(hr, s * 32, j0, lane);
      v16bf bhi = load_bT(hi, s * 32, j0, lane);
      v1 = wmma_bf16(amr[s], bhr, v1);   // Mr * hr
      v2 = wmma_bf16(ami[s], bhr, v2);   // Mi * hr
      v3 = wmma_bf16(amr[s], bhi, v3);   // Mr * hi
      v4 = wmma_bf16(ami[s], bhi, v4);   // Mi * hi
    }
    // diag-dot: lane's 8 values Mr[k0+mh+v][j0+n] are contiguous in MrT[j0+n][k0+mh + v]
    v8bf mrv = *(const v8bf*)(MrT + (size_t)(j0 + n) * DD + k0 + mh);   // 16B aligned
    v8bf miv = *(const v8bf*)(MiT + (size_t)(j0 + n) * DD + k0 + mh);
#pragma unroll
    for (int v = 0; v < 8; ++v) {
      float mr = (float)mrv[v];
      float mi = (float)miv[v];
      ps[v] += v1[v] * mr + v2[v] * mi - v3[v] * mi + v4[v] * mr;
    }
  }
  __shared__ float red[32][8];
#pragma unroll
  for (int v = 0; v < 8; ++v) red[lane][v] = ps[v];
  __syncthreads();
  if (lane < 16) {
    int base = (lane < 8) ? 0 : 16;   // row m<8 lives in lanes 0-15, m>=8 in lanes 16-31
    int v = lane & 7;
    float s = 0.f;
    for (int l = 0; l < 16; ++l) s += red[base + l][v];
    // fold final QActivation analytically: probs(y2) = BETA*probs(n2) + (1-BETA)/D
    probs[((size_t)t * BB + b) * DD + k0 + lane] = BETA_C * s + (1.f - BETA_C) / (float)DD;
  }
}

// ---------------- small VALU kernels ----------------

__global__ __launch_bounds__(256)
void cvtU_k(const float* a, const float* b, const float* c, const float* d,
            __bf16* A, __bf16* B, __bf16* C, __bf16* Dp) {
  int i = blockIdx.x * 256 + threadIdx.x;
  if (i < DD * DD) { A[i] = (__bf16)a[i]; B[i] = (__bf16)b[i]; C[i] = (__bf16)c[i]; Dp[i] = (__bf16)d[i]; }
}

__global__ __launch_bounds__(256)
void normM_k(const float* Kr, const float* Ki,
             __bf16* Mr, __bf16* Mi, __bf16* MrT, __bf16* MiT) {
  int k = threadIdx.x;
  if (k >= DD) return;
  const float* kr = Kr + (size_t)k * DD;
  const float* ki = Ki + (size_t)k * DD;
  float ss = 0.f;
  for (int j = 0; j < DD; ++j) ss += kr[j] * kr[j] + ki[j] * ki[j];
  float inv = 1.f / sqrtf(ss);
  for (int j = 0; j < DD; ++j) {
    __bf16 vr = (__bf16)(kr[j] * inv);
    __bf16 vi = (__bf16)(ki[j] * inv);
    Mr[(size_t)k * DD + j] = vr;
    Mi[(size_t)k * DD + j] = vi;
    MrT[(size_t)j * DD + k] = vr;     // transposed copy for contiguous diag-dot reads
    MiT[(size_t)j * DD + k] = vi;
  }
}

template<int FI, int FO>
__device__ inline void pure_state_dev(const float* x, const float* W, const float* bias,
                                      const float* P, int spk, float* outr, float* outi) {
  float r[FO]; float ss = 0.f;
  for (int o = 0; o < FO; ++o) {
    float acc = bias[o];
    for (int i = 0; i < FI; ++i) acc = fmaf(x[i], W[i * FO + o], acc);
    acc = fmaxf(acc, 0.f);
    r[o] = acc; ss += acc * acc;
  }
  float nrm = fmaxf(sqrtf(ss), 1e-12f);
  for (int o = 0; o < FO; ++o) {
    float amp = r[o] / nrm;
    float ph = P[spk * FO + o];
    outr[o] = amp * cosf(ph);
    outi[o] = amp * sinf(ph);
  }
}

__global__ __launch_bounds__(256)
void front_k(const float* xt, const float* xa, const float* xv, const float* sm,
             const float* Wt, const float* bt, const float* Wa, const float* ba,
             const float* Wv, const float* bv,
             const float* Pt, const float* Pa, const float* Pv,
             float* psir, float* psii) {
  int idx = threadIdx.x;           // one block of 256 = all (b,t)
  int b = idx >> 5, t = idx & 31;
  size_t row = (size_t)b * TT + t;
  const float* s2 = sm + row * 2;
  int spk = (s2[1] > s2[0]) ? 1 : 0;
  float trr[4], tii[4], arr[8], aii[8], vrr[8], vii[8];
  pure_state_dev<300, 4>(xt + row * 300, Wt, bt, Pt, spk, trr, tii);
  pure_state_dev<74, 8>(xa + row * 74, Wa, ba, Pa, spk, arr, aii);
  pure_state_dev<35, 8>(xv + row * 35, Wv, bv, Pv, spk, vrr, vii);
  float* pr = psir + row * DD;
  float* pi = psii + row * DD;
  for (int it = 0; it < 4; ++it)
    for (int ia = 0; ia < 8; ++ia) {
      float qr = trr[it] * arr[ia] - tii[it] * aii[ia];
      float qi = trr[it] * aii[ia] + tii[it] * arr[ia];
      int base = (it * 8 + ia) * 8;
      for (int iv = 0; iv < 8; ++iv) {
        pr[base + iv] = qr * vrr[iv] - qi * vii[iv];
        pi[base + iv] = qr * vii[iv] + qi * vrr[iv];
      }
    }
}

__global__ __launch_bounds__(256)
void initH_k(__bf16* h1tr, __bf16* h1ti, __bf16* h2tr, __bf16* h2ti) {
  size_t idx = (size_t)blockIdx.x * 256 + threadIdx.x;
  if (idx >= (size_t)BB * DD * DD) return;
  int ij = (int)(idx & (DD * DD - 1));
  int i = ij >> 8, j = ij & 255;
  __bf16 v = (__bf16)((i == j) ? (1.f / DD) : 0.f);   // identity: transpose-invariant
  h1tr[idx] = v; h1ti[idx] = (__bf16)0.f;
  h2tr[idx] = v; h2ti[idx] = (__bf16)0.f;
}

// layer-1 combine: n1 = lam*(phi phi^dag) + (1-lam)*he1 ; store h1,y1 TRANSPOSED in bf16
__global__ __launch_bounds__(256)
void combine1_k(const float* __restrict__ her, const float* __restrict__ hei,
                const float* __restrict__ phr, const float* __restrict__ phiv,
                const float* __restrict__ Lambda,
                __bf16* __restrict__ h1tr, __bf16* __restrict__ h1ti,
                __bf16* __restrict__ y1tr, __bf16* __restrict__ y1ti, int t) {
  size_t idx = (size_t)blockIdx.x * 256 + threadIdx.x;
  if (idx >= (size_t)BB * DD * DD) return;
  int b  = (int)(idx >> 16);
  int ij = (int)(idx & 65535);
  int i = ij >> 8, j = ij & 255;
  float lam = 1.f / (1.f + expf(-Lambda[0]));
  const float* pr = phr  + ((size_t)b * TT + t) * DD;
  const float* pi = phiv + ((size_t)b * TT + t) * DD;
  float pri = pr[i], prj = pr[j], pii = pi[i], pij = pi[j];
  float xer = pri * prj + pii * pij;         // rank-1 evolved rho (real)
  float xei = pii * prj - pri * pij;         // rank-1 evolved rho (imag)
  float nr = lam * xer + (1.f - lam) * her[idx];
  float ni = lam * xei + (1.f - lam) * hei[idx];
  size_t tix = (size_t)b * DD * DD + (size_t)j * DD + i;   // transposed store
  h1tr[tix] = (__bf16)nr;
  h1ti[tix] = (__bf16)ni;
  y1tr[tix] = (__bf16)(BETA_C * nr + ((i == j) ? (1.f - BETA_C) / (float)DD : 0.f));
  y1ti[tix] = (__bf16)(BETA_C * ni);
}

__global__ __launch_bounds__(256)
void combine2_k(const float* __restrict__ xer, const float* __restrict__ xei,
                const float* __restrict__ her, const float* __restrict__ hei,
                const float* __restrict__ Lambda,
                __bf16* __restrict__ h2tr, __bf16* __restrict__ h2ti) {
  size_t idx = (size_t)blockIdx.x * 256 + threadIdx.x;
  if (idx >= (size_t)BB * DD * DD) return;
  int b  = (int)(idx >> 16);
  int ij = (int)(idx & 65535);
  int i = ij >> 8, j = ij & 255;
  float lam = 1.f / (1.f + expf(-Lambda[0]));
  size_t tix = (size_t)b * DD * DD + (size_t)j * DD + i;   // transposed store
  h2tr[tix] = (__bf16)(lam * xer[idx] + (1.f - lam) * her[idx]);
  h2ti[tix] = (__bf16)(lam * xei[idx] + (1.f - lam) * hei[idx]);
}

__global__ __launch_bounds__(256)
void head_k(const float* probs, const float* W1, const float* b1,
            const float* W2, const float* b2, float* out) {
  int idx = threadIdx.x;           // one block of 256 = all (b,t)
  int b = idx >> 5, t = idx & 31;
  const float* p = probs + ((size_t)t * BB + b) * DD;
  float acc[6];
  for (int c = 0; c < 6; ++c) acc[c] = b2[c];
  for (int o = 0; o < 64; ++o) {
    float h = b1[o];
    for (int k = 0; k < DD; ++k) h = fmaf(p[k], W1[(size_t)k * 64 + o], h);
    h = fmaxf(h, 0.f);
    for (int c = 0; c < 6; ++c) acc[c] = fmaf(h, W2[o * 6 + c], acc[c]);
  }
  float v[6]; float mx = -1e30f;
  for (int c = 0; c < 6; ++c) { v[c] = tanhf(acc[c]); mx = fmaxf(mx, v[c]); }
  float s = 0.f;
  for (int c = 0; c < 6; ++c) s += expf(v[c] - mx);
  float lse = logf(s);
  float* o = out + ((size_t)b * TT + t) * 6;
  for (int c = 0; c < 6; ++c) o[c] = v[c] - mx - lse;
}

// ---------------- host launcher ----------------
extern "C" void kernel_launch(void* const* d_in, const int* in_sizes, int n_in,
                              void* d_out, int out_size, void* d_ws, size_t ws_size,
                              hipStream_t stream) {
  const float* x_text  = (const float*)d_in[0];
  const float* x_audio = (const float*)d_in[1];
  const float* x_video = (const float*)d_in[2];
  const float* smask   = (const float*)d_in[3];
  const float* Wt = (const float*)d_in[4];
  const float* bt = (const float*)d_in[5];
  const float* Wa = (const float*)d_in[6];
  const float* ba = (const float*)d_in[7];
  const float* Wv = (const float*)d_in[8];
  const float* bv = (const float*)d_in[9];
  const float* Pt = (const float*)d_in[10];
  const float* Pa = (const float*)d_in[11];
  const float* Pv = (const float*)d_in[12];
  const float* Uxr = (const float*)d_in[13];
  const float* Uxi = (const float*)d_in[14];
  const float* Uhr = (const float*)d_in[15];
  const float* Uhi = (const float*)d_in[16];
  const float* Lambda = (const float*)d_in[17];
  const float* Kr = (const float*)d_in[18];
  const float* Ki = (const float*)d_in[19];
  const float* W1 = (const float*)d_in[20];
  const float* b1 = (const float*)d_in[21];
  const float* W2 = (const float*)d_in[22];
  const float* b2 = (const float*)d_in[23];
  float* out = (float*)d_out;
  (void)in_sizes; (void)n_in; (void)out_size; (void)ws_size;

  char* w = (char*)d_ws;
  auto carve = [&](size_t bytes) -> void* {
    char* p = w; w += (bytes + 255) & ~(size_t)255; return (void*)p;
  };
  const size_t MM  = (size_t)DD * DD;     // 65536
  const size_t BMM = (size_t)BB * MM;     // 524288
  __bf16* bUxr = (__bf16*)carve(MM * 2);
  __bf16* bUxi = (__bf16*)carve(MM * 2);
  __bf16* bUhr = (__bf16*)carve(MM * 2);
  __bf16* bUhi = (__bf16*)carve(MM * 2);
  __bf16* bMr  = (__bf16*)carve(MM * 2);
  __bf16* bMi  = (__bf16*)carve(MM * 2);
  __bf16* bMrT = (__bf16*)carve(MM * 2);
  __bf16* bMiT = (__bf16*)carve(MM * 2);
  float* psir = (float*)carve(MM * 4);    // (B*T, D) rows = b*T + t
  float* psii = (float*)carve(MM * 4);
  float* phr  = (float*)carve(MM * 4);    // Phi = Ux psi (all rows at once)
  float* phiv = (float*)carve(MM * 4);
  __bf16* h1tr = (__bf16*)carve(BMM * 2); // transposed bf16 states
  __bf16* h1ti = (__bf16*)carve(BMM * 2);
  __bf16* h2tr = (__bf16*)carve(BMM * 2);
  __bf16* h2ti = (__bf16*)carve(BMM * 2);
  __bf16* y1tr = (__bf16*)carve(BMM * 2);
  __bf16* y1ti = (__bf16*)carve(BMM * 2);
  float* sAr = (float*)carve(BMM * 4);    // f32 intermediates (A-operand of RIGHT pass)
  float* sAi = (float*)carve(BMM * 4);
  float* sBr = (float*)carve(BMM * 4);
  float* sBi = (float*)carve(BMM * 4);
  float* sCr = (float*)carve(BMM * 4);
  float* sCi = (float*)carve(BMM * 4);
  float* probs = (float*)carve((size_t)TT * BB * DD * 4);

  dim3 blk32(32, 1, 1);
  dim3 gU(4, 16, 1);
  dim3 gB(4, 16, BB);
  dim3 gM(16, BB, 1);
  const int nElem = (int)((BMM + 255) / 256);

  cvtU_k<<<DD, 256, 0, stream>>>(Uxr, Uxi, Uhr, Uhi, bUxr, bUxi, bUhr, bUhi);
  normM_k<<<1, 256, 0, stream>>>(Kr, Ki, bMr, bMi, bMrT, bMiT);
  front_k<<<1, 256, 0, stream>>>(x_text, x_audio, x_video, smask,
                                 Wt, bt, Wa, ba, Wv, bv, Pt, Pa, Pv, psir, psii);
  // Phi = psi * Ux^T (complex, no conj) == (Ux psi) per row; layer-1 x-evolve is rank-1
  cgemm256<float, false><<<gU, blk32, 0, stream>>>(
      psir, psii, 0, bUxr, bUxi, 0, phr, phiv, 0);
  initH_k<<<nElem, 256, 0, stream>>>(h1tr, h1ti, h2tr, h2ti);

  for (int t = 0; t < TT; ++t) {
    // he1 = Uh h1 Uh^dagger  (h1 stored transposed -> contiguous B frags)
    cgemm256<__bf16, false><<<gB, blk32, 0, stream>>>(
        bUhr, bUhi, 0, h1tr, h1ti, (long)MM, sAr, sAi, (long)MM);
    cgemm256<float, true><<<gB, blk32, 0, stream>>>(
        sAr, sAi, (long)MM, bUhr, bUhi, 0, sCr, sCi, (long)MM);
    combine1_k<<<nElem, 256, 0, stream>>>(sCr, sCi, phr, phiv, Lambda,
                                          h1tr, h1ti, y1tr, y1ti, t);
    // xe2 = Ux y1 Ux^dagger
    cgemm256<__bf16, false><<<gB, blk32, 0, stream>>>(
        bUxr, bUxi, 0, y1tr, y1ti, (long)MM, sAr, sAi, (long)MM);
    cgemm256<float, true><<<gB, blk32, 0, stream>>>(
        sAr, sAi, (long)MM, bUxr, bUxi, 0, sBr, sBi, (long)MM);
    // he2 = Uh h2 Uh^dagger
    cgemm256<__bf16, false><<<gB, blk32, 0, stream>>>(
        bUhr, bUhi, 0, h2tr, h2ti, (long)MM, sAr, sAi, (long)MM);
    cgemm256<float, true><<<gB, blk32, 0, stream>>>(
        sAr, sAi, (long)MM, bUhr, bUhi, 0, sCr, sCi, (long)MM);
    combine2_k<<<nElem, 256, 0, stream>>>(sBr, sBi, sCr, sCi, Lambda, h2tr, h2ti);
    // measurement of layer-2 state (activation folded analytically)
    meas_k<<<gM, blk32, 0, stream>>>(h2tr, h2ti, bMr, bMi, bMrT, bMiT, probs, t);
  }
  head_k<<<1, 256, 0, stream>>>(probs, W1, b1, W2, b2, out);
}
